// GeneratedTreeClassifier_15522011808539
// MI455X (gfx1250) — matrix operand order repaired and checked by reading, not compiled, not run on hardware
//
#include <hip/hip_runtime.h>
#include <hip/hip_bf16.h>
#include <math.h>

// ---------------- problem constants ----------------
#define D_IN     2048
#define N_TREES  15
#define N_INT    7
#define N_LEAF   8
#define PPT      (N_INT * (D_IN + 1) + N_LEAF * 2)   // 14359 params per tree
#define NCOL     (N_TREES * N_INT)                   // 105 decision columns
#define NTILE    7                                   // 7 x 16 = 112 padded cols
#define NPAD     (NTILE * 16)
#define KCHUNK   32                                  // WMMA bf16 K
#define KCHUNKS  (D_IN / KCHUNK)                     // 64
#define KPAIRS   (KCHUNKS / 2)                       // 32 double-chunks
#define M_BLK    128                                 // rows per workgroup (8 waves x 16)
#define BATCH_N  32768
#define DEC_STRIDE 113                               // LDS row stride (avoid bank conflicts)

// packed-W geometry (dwords)
#define FRAG_DW   1792                    // one term (hi or lo): 7 tiles * 32 lanes * 8 dw
#define KC_DW     (2 * FRAG_DW)           // one K-chunk: hi + lo            = 3584 dw
#define KK_DW     (2 * KC_DW)             // one K-pair (2 chunks)           = 7168 dw
#define KK_BYTES  (KK_DW * 4)             // 28672 B = 7 * 256 threads * 16 B
#define WPK_DW    (KCHUNKS * KC_DW)       // 229376 dw total

typedef __attribute__((ext_vector_type(16))) __bf16        v16bf;
typedef __attribute__((ext_vector_type(8)))  float         v8f;
typedef __attribute__((ext_vector_type(8)))  unsigned int  v8u;

// round-to-nearest-even f32 -> bf16 bits (scalar fallback)
static __device__ __forceinline__ unsigned int bf16rn(float a) {
    unsigned int u = __float_as_uint(a);
    u += 0x7FFFu + ((u >> 16) & 1u);
    return u >> 16;
}
static __device__ __forceinline__ float bf2f(unsigned int h) {
    return __uint_as_float(h << 16);
}
// pack two f32 -> two bf16 in one dword (a -> low, b -> high)
static __device__ __forceinline__ unsigned int cvt_pk_bf16(float a, float b) {
#if __has_builtin(__builtin_amdgcn_cvt_pk_bf16_f32)
    typedef __attribute__((ext_vector_type(2))) __bf16 v2bf;
    v2bf r = __builtin_amdgcn_cvt_pk_bf16_f32(a, b);
    return __builtin_bit_cast(unsigned int, r);
#else
    return bf16rn(a) | (bf16rn(b) << 16);
#endif
}

// =====================================================================
// Pack split_w into WMMA B-fragment-major hi/lo bf16 buffers.
// Layout (dwords): [kc][term(hi=0,lo=1)][nt][lane][v0..7]
// B fragment k order chosen here (contiguous, matches A; bf16 32x16, wave32): lanes 0-15 cover K 0..15,
// lanes 16-31 cover K 16..31, dword v packs (k_local = khalf*16+2v, +1),
// n = nt*16 + (lane&15).
// =====================================================================
__global__ __launch_bounds__(256) void pack_w_kernel(
    const float* __restrict__ params, unsigned int* __restrict__ wpk) {
    int idx = blockIdx.x * 256 + threadIdx.x;           // < KCHUNKS*NTILE*32*8
    int v    = idx & 7;
    int lane = (idx >> 3) & 31;
    int nt   = (idx >> 8) % NTILE;
    int kc   = idx / (8 * 32 * NTILE);
    int k0   = kc * KCHUNK + (lane >> 4) * 16 + 2 * v;
    int n    = nt * 16 + (lane & 15);

    float w0 = 0.f, w1 = 0.f;
    if (n < NCOL) {
        int t = n / N_INT, i = n % N_INT;
        const float* wr = params + t * PPT + i * D_IN;
        w0 = wr[k0];
        w1 = wr[k0 + 1];
    }
    unsigned int h0 = bf16rn(w0), h1 = bf16rn(w1);
    unsigned int l0 = bf16rn(w0 - bf2f(h0)), l1 = bf16rn(w1 - bf2f(h1));
    int frag = ((nt * 32 + lane) << 3) + v;             // 0..1791
    wpk[kc * KC_DW + frag]           = h0 | (h1 << 16); // hi term
    wpk[kc * KC_DW + FRAG_DW + frag] = l0 | (l1 << 16); // lo term
}

// Pack biases (padded), softmaxed leaf distributions, tree weights.
__global__ __launch_bounds__(256) void pack_misc_kernel(
    const float* __restrict__ params, const float* __restrict__ tweights,
    float* __restrict__ bias, float* __restrict__ leafp, float* __restrict__ tw) {
    int tid = threadIdx.x;
    if (tid < NPAD) {
        float b = 0.f;
        if (tid < NCOL) {
            int t = tid / N_INT, i = tid % N_INT;
            b = params[t * PPT + N_INT * D_IN + i];
        }
        bias[tid] = b;
    }
    if (tid < N_TREES * N_LEAF) {
        int t = tid / N_LEAF, l = tid % N_LEAF;
        const float* ll = params + t * PPT + N_INT * (D_IN + 1) + l * 2;
        float a = ll[0], b = ll[1];
        float m = fmaxf(a, b);
        float ea = expf(a - m), eb = expf(b - m);
        float inv = 1.f / (ea + eb);
        leafp[tid * 2 + 0] = ea * inv;
        leafp[tid * 2 + 1] = eb * inv;
    }
    if (tid < N_TREES) tw[tid] = tweights[tid];
}

// =====================================================================
// Main fused kernel: split-precision bf16 WMMA GEMM + soft-tree epilogue.
// 8 waves/block; wave w owns rows [blk*128 + w*16, +16), all 112 cols.
// B fragments are double-buffered in LDS via global_load_async_to_lds_b128.
// =====================================================================
__global__ __launch_bounds__(256) void tree_fwd_kernel(
    const float* __restrict__ x, const unsigned int* __restrict__ wpk,
    const float* __restrict__ bias, const float* __restrict__ leafp,
    const float* __restrict__ tw, float* __restrict__ out) {

    // staging (2 x 28672 B, alive during K-loop) aliased with the decision
    // buffer (57856 B, alive after the K-loop).
    __shared__ unsigned char smem_raw[M_BLK * DEC_STRIDE * 4] __attribute__((aligned(16)));
    const unsigned int* bsm = reinterpret_cast<const unsigned int*>(smem_raw);
    float* dec = reinterpret_cast<float*>(smem_raw);

    const int tid  = threadIdx.x;
    const int wave = tid >> 5;
    const int lane = tid & 31;
    const int half = lane >> 4;
    const int rowBlock = blockIdx.x * M_BLK;
    const int aRow = rowBlock + wave * 16 + (lane & 15);
    const float* __restrict__ xrow = x + (size_t)aRow * D_IN;
    const unsigned ldsbase = (unsigned)(uintptr_t)(void*)smem_raw;

    // async-stage one K-pair (28672 B) of packed W into LDS buffer `buf`
    auto stage = [&](int kk, int buf) {
        const unsigned char* gp0 = (const unsigned char*)wpk + (size_t)kk * KK_BYTES;
        unsigned lb = ldsbase + (unsigned)buf * KK_BYTES;
#pragma unroll
        for (int j = 0; j < 7; ++j) {
            unsigned off = (unsigned)(j * 256 + tid) * 16u;
            unsigned la = lb + off;
            const unsigned char* ga = gp0 + off;
            asm volatile("global_load_async_to_lds_b128 %0, %1, off"
                         :: "v"(la), "v"(ga) : "memory");
        }
    };

    v8f acc[NTILE];
#pragma unroll
    for (int nt = 0; nt < NTILE; ++nt) acc[nt] = (v8f)0.f;

    stage(0, 0);
    for (int kk = 0; kk < KPAIRS; ++kk) {
        if (kk + 1 < KPAIRS) {
            stage(kk + 1, (kk + 1) & 1);
            asm volatile("s_wait_asynccnt 0x7" ::: "memory");   // kk's fills done
        } else {
            asm volatile("s_wait_asynccnt 0x0" ::: "memory");
        }
        __syncthreads();                                        // all waves filled

        // prefetch next K-pair of X rows
        if (kk + 1 < KPAIRS) {
            __builtin_prefetch(xrow + (kk + 1) * 2 * KCHUNK, 0, 0);
            __builtin_prefetch(xrow + (kk + 1) * 2 * KCHUNK + 32, 0, 0);
        }

        const unsigned int* bb = bsm + (kk & 1) * KK_DW;
#pragma unroll
        for (int p = 0; p < 2; ++p) {
            const int kc = kk * 2 + p;
            const float* xk = xrow + kc * KCHUNK;

            // ---- A fragment: 16 f32 -> bf16 hi/lo, contiguous k order (same as B) ----
            // lanes 0-15: v0-3 = K 0..7,  v4-7 = K 8..15
            // lanes16-31: v0-3 = K 16..23, v4-7 = K 24..31
            const float4* xf = reinterpret_cast<const float4*>(xk + half * 16);
            const float4* xg = reinterpret_cast<const float4*>(xk + half * 16 + 8);
            float4 f01 = xf[0], f23 = xf[1];
            float4 g01 = xg[0], g23 = xg[1];
            float fv[8] = {f01.x, f01.y, f01.z, f01.w, f23.x, f23.y, f23.z, f23.w};
            float gv[8] = {g01.x, g01.y, g01.z, g01.w, g23.x, g23.y, g23.z, g23.w};

            v8u ahi_u, alo_u;
#pragma unroll
            for (int v = 0; v < 4; ++v) {
                float a0 = fv[2 * v], a1 = fv[2 * v + 1];
                unsigned hp = cvt_pk_bf16(a0, a1);
                ahi_u[v] = hp;
                float r0 = a0 - __uint_as_float(hp << 16);
                float r1 = a1 - __uint_as_float(hp & 0xFFFF0000u);
                alo_u[v] = cvt_pk_bf16(r0, r1);

                float b0 = gv[2 * v], b1 = gv[2 * v + 1];
                unsigned hq = cvt_pk_bf16(b0, b1);
                ahi_u[v + 4] = hq;
                float s0 = b0 - __uint_as_float(hq << 16);
                float s1 = b1 - __uint_as_float(hq & 0xFFFF0000u);
                alo_u[v + 4] = cvt_pk_bf16(s0, s1);
            }
            v16bf Ahi = __builtin_bit_cast(v16bf, ahi_u);
            v16bf Alo = __builtin_bit_cast(v16bf, alo_u);

            // ---- 7 column tiles: B hi/lo fragments from LDS, 3 WMMAs each ----
            const unsigned int* bh_base = bb + p * KC_DW + ((lane) << 3);
#pragma unroll
            for (int nt = 0; nt < NTILE; ++nt) {
                const uint4* bh = reinterpret_cast<const uint4*>(bh_base + nt * 256);
                const uint4* bl = reinterpret_cast<const uint4*>(bh_base + FRAG_DW + nt * 256);
                uint4 h0 = bh[0], h1 = bh[1];
                uint4 l0 = bl[0], l1 = bl[1];
                v8u bhu = {h0.x, h0.y, h0.z, h0.w, h1.x, h1.y, h1.z, h1.w};
                v8u blu = {l0.x, l0.y, l0.z, l0.w, l1.x, l1.y, l1.z, l1.w};
                v16bf Bhi = __builtin_bit_cast(v16bf, bhu);
                v16bf Blo = __builtin_bit_cast(v16bf, blu);

                acc[nt] = __builtin_amdgcn_wmma_f32_16x16x32_bf16(
                    false, Ahi, false, Bhi, (short)0, acc[nt], false, false);
                acc[nt] = __builtin_amdgcn_wmma_f32_16x16x32_bf16(
                    false, Ahi, false, Blo, (short)0, acc[nt], false, false);
                acc[nt] = __builtin_amdgcn_wmma_f32_16x16x32_bf16(
                    false, Alo, false, Bhi, (short)0, acc[nt], false, false);
            }
        }
        __syncthreads();   // everyone done reading buf[kk&1] before refill
    }

    // ---- transpose accumulators into LDS: dec[row_local][col] ----
    // C/D layout: n = lane&15; VGPR j -> M = j + 8*half
#pragma unroll
    for (int nt = 0; nt < NTILE; ++nt) {
        int col = nt * 16 + (lane & 15);
#pragma unroll
        for (int j = 0; j < 8; ++j) {
            int row_local = wave * 16 + j + 8 * half;
            dec[row_local * DEC_STRIDE + col] = acc[nt][j];
        }
    }
    __syncthreads();

    // ---- soft-tree epilogue: one thread per row ----
    int row_local = tid;
    if (row_local < M_BLK) {
        int row = rowBlock + row_local;
        const float* drow = dec + row_local * DEC_STRIDE;
        float o0 = 0.f, o1 = 0.f;
#pragma unroll 1
        for (int t = 0; t < N_TREES; ++t) {
            float d[N_INT];
#pragma unroll
            for (int i = 0; i < N_INT; ++i) {
                float z = drow[t * N_INT + i] + bias[t * N_INT + i];
                d[i] = 1.f / (1.f + expf(-z));
            }
            float p[N_LEAF];
#pragma unroll
            for (int l = 0; l < N_LEAF; ++l) p[l] = 1.f;
#pragma unroll
            for (int i = 0; i < N_INT; ++i) {
                int left = 2 * i + 1, right = 2 * i + 2;
                if (left < N_LEAF) p[left] *= d[i];
                if (right < N_LEAF) p[right] *= (1.f - d[i]);
            }
            float s = 0.f;
#pragma unroll
            for (int l = 0; l < N_LEAF; ++l) s += p[l];
            float inv = 1.f / (s + 1e-8f);
            float wt = tw[t];
#pragma unroll
            for (int l = 0; l < N_LEAF; ++l) {
                float pl = p[l] * inv * wt;
                o0 += pl * leafp[(t * N_LEAF + l) * 2 + 0];
                o1 += pl * leafp[(t * N_LEAF + l) * 2 + 1];
            }
        }
        reinterpret_cast<float2*>(out)[row] = make_float2(o0, o1);
    }
}

// =====================================================================
extern "C" void kernel_launch(void* const* d_in, const int* in_sizes, int n_in,
                              void* d_out, int out_size, void* d_ws, size_t ws_size,
                              hipStream_t stream) {
    const float* x        = (const float*)d_in[0];
    const float* params   = (const float*)d_in[1];
    const float* tweights = (const float*)d_in[2];
    float* out            = (float*)d_out;

    unsigned int* wpk = (unsigned int*)d_ws;
    float* bias  = (float*)(wpk + WPK_DW);
    float* leafp = bias + NPAD;
    float* tw    = leafp + N_TREES * N_LEAF * 2;

    pack_w_kernel<<<(KCHUNKS * NTILE * 32 * 8) / 256, 256, 0, stream>>>(params, wpk);
    pack_misc_kernel<<<1, 256, 0, stream>>>(params, tweights, bias, leafp, tw);
    tree_fwd_kernel<<<BATCH_N / M_BLK, 256, 0, stream>>>(
        x, wpk, bias, leafp, tw, out);
}
